// TransformerBlock_61813169324160
// MI455X (gfx1250) — matrix operand order repaired and checked by reading, not compiled, not run on hardware
//
#include <hip/hip_runtime.h>
#include <math.h>

// ---------------------------------------------------------------------------
// Problem constants (from the reference)
// ---------------------------------------------------------------------------
#define BATCH  8
#define DIMC   192
#define HEADS  6
#define HH     128
#define WW2    128
#define NPIX   (HH * WW2)          // 16384 pixels per batch image
#define QKVC   (3 * DIMC)          // 576
#define HD     (DIMC / HEADS)      // 32
#define H1C    96
#define H2C    96

typedef __attribute__((ext_vector_type(16))) __bf16 v16bf;
typedef __attribute__((ext_vector_type(8)))  float  v8f;

// ---------------------------------------------------------------------------
// bf16 <-> f32 helpers (bit-exact RNE, no reliance on __bf16 arith support)
// ---------------------------------------------------------------------------
__device__ __forceinline__ __bf16 to_bf(float f) {
    unsigned u = __float_as_uint(f);
    u += 0x7fffu + ((u >> 16) & 1u);          // round-to-nearest-even
    unsigned short h = (unsigned short)(u >> 16);
    __bf16 r;
    __builtin_memcpy(&r, &h, 2);
    return r;
}
__device__ __forceinline__ float from_bf(__bf16 b) {
    unsigned short h;
    __builtin_memcpy(&h, &b, 2);
    return __uint_as_float(((unsigned)h) << 16);
}
__device__ __forceinline__ float gelu_f(float x) {
    return 0.5f * x * (1.0f + erff(x * 0.70710678118654752f));
}
__device__ __forceinline__ float sigmoid_f(float x) {
    return 1.0f / (1.0f + expf(-x));
}

// ---------------------------------------------------------------------------
// Fold BN (g,b,m,v) into per-channel scale/shift: s = g/sqrt(v+eps), t = b-m*s
// ---------------------------------------------------------------------------
__global__ void bn_prep(const float* __restrict__ g, const float* __restrict__ b,
                        const float* __restrict__ m, const float* __restrict__ v,
                        float* __restrict__ s, float* __restrict__ t, int C) {
    int c = blockIdx.x * blockDim.x + threadIdx.x;
    if (c >= C) return;
    float sc = g[c] * rsqrtf(v[c] + 1e-5f);
    s[c] = sc;
    t[c] = b[c] - m[c] * sc;
}

// ---------------------------------------------------------------------------
// Repack 1x1-conv weight [CO][CI] (f32) into WMMA A-operand tile layout:
// out[(coT*(KP/32)+kt)*512 + lane*16 + e]  (bf16), zero-padded for k >= CI.
// Contiguous-k A 16x32 layout used here (B matches): lane m = lane&15, half = lane>>4,
//   element e maps to K = e + 16*half.
// ---------------------------------------------------------------------------
__global__ void pack_w(const float* __restrict__ W, __bf16* __restrict__ out,
                       int CO, int CI, int KP) {
    int tid   = blockIdx.x * blockDim.x + threadIdx.x;
    int total = (CO / 16) * (KP / 32) * 512;
    if (tid >= total) return;
    int tile = tid >> 9;
    int r    = tid & 511;
    int lane = r >> 4;
    int e    = r & 15;
    int m    = lane & 15;
    int half = lane >> 4;
    int K    = e + 16 * half;
    int ktiles = KP >> 5;
    int kt = tile % ktiles;
    int ct = tile / ktiles;
    int co = ct * 16 + m;
    int k  = kt * 32 + K;
    float v = (k < CI) ? W[(size_t)co * CI + k] : 0.0f;
    out[tid] = to_bf(v);
}

// ---------------------------------------------------------------------------
// Channel LayerNorm: x f32 [B][192][N]  ->  y bf16 pixel-major [B][N][192]
// One thread per pixel; channel loads are coalesced across the block.
// ---------------------------------------------------------------------------
__global__ void ln_kernel(const float* __restrict__ x, const float* __restrict__ g,
                          const float* __restrict__ beta, __bf16* __restrict__ y) {
    size_t pid = (size_t)blockIdx.x * blockDim.x + threadIdx.x;
    if (pid >= (size_t)BATCH * NPIX) return;
    size_t b = pid / NPIX, p = pid % NPIX;
    const float* xb = x + (b * DIMC) * (size_t)NPIX + p;
    float s0 = 0.f, s1 = 0.f;
    for (int c = 0; c < DIMC; ++c) {
        float v = xb[(size_t)c * NPIX];
        s0 += v; s1 += v * v;
    }
    float mu  = s0 * (1.0f / DIMC);
    float var = s1 * (1.0f / DIMC) - mu * mu;
    float inv = rsqrtf(var + 1e-5f);
    __bf16* yo = y + pid * DIMC;
    for (int c = 0; c < DIMC; ++c) {
        float v = (xb[(size_t)c * NPIX] - mu) * inv * g[c] + beta[c];
        yo[c] = to_bf(v);
    }
}

// ---------------------------------------------------------------------------
// Generic WMMA GEMM:  OUT[b][co][p] = sum_k Wpack[co][k] * Bact[b][p][ci_off+k]
// Each wave owns MT consecutive 16-co tiles (16 pixels x MT*16 channels):
// one B-tile fetch feeds MT WMMAs, cutting activation traffic by MT.
// K-loop body: gather B + all MT A tiles first (wide load front, partial
// loadcnt waits), then issue the MT WMMAs. Per-i A row pointers are hoisted
// so the K-step is a constant byte offset.
// Epilogue: optional BN (s,t), optional exact GELU, optional residual add,
//           outputs: f32 ch-major / bf16 ch-major / bf16 pixel-major (stride CBo)
// Block = 256 threads = 8 waves; grid = (N/128, CO/(16*MT), B).
// ---------------------------------------------------------------------------
template <int MT>
__global__ void gemm_wmma(const __bf16* __restrict__ Apack,
                          const __bf16* __restrict__ Bact, int CB, int ci_off,
                          int KP, int CO,
                          const float* __restrict__ s, const float* __restrict__ t,
                          int act,
                          const float* __restrict__ res,
                          float* __restrict__ outf,
                          __bf16* __restrict__ outbc,
                          __bf16* __restrict__ outb, int CBo) {
    const int wave = threadIdx.x >> 5;
    const int lane = threadIdx.x & 31;
    const int nT   = blockIdx.x * 8 + wave;   // pixel tile
    const int coT  = blockIdx.y;              // group of MT channel tiles
    const int b    = blockIdx.z;
    const int half = lane >> 4;
    const int p    = nT * 16 + (lane & 15);
    const int ktiles = KP >> 5;

    v8f acc[MT];
#pragma unroll
    for (int i = 0; i < MT; ++i) acc[i] = v8f{};

    const __bf16* brow = Bact + ((size_t)b * NPIX + p) * CB + ci_off + 16 * half;
    const __bf16* arows[MT];
#pragma unroll
    for (int i = 0; i < MT; ++i)
        arows[i] = Apack + ((size_t)(coT * MT + i) * ktiles) * 512 + lane * 16;

    for (int kt = 0; kt < ktiles; ++kt) {
        v16bf bb = *(const v16bf*)(brow + kt * 32);
        v16bf a[MT];
#pragma unroll
        for (int i = 0; i < MT; ++i)
            a[i] = *(const v16bf*)(arows[i] + (size_t)kt * 512);
        if (kt + 1 < ktiles)
            __builtin_prefetch(brow + (kt + 1) * 32, 0, 1);   // global_prefetch
#pragma unroll
        for (int i = 0; i < MT; ++i)
            acc[i] = __builtin_amdgcn_wmma_f32_16x16x32_bf16(
                         false, a[i], false, bb, (short)0, acc[i], false, false);
    }

#pragma unroll
    for (int i = 0; i < MT; ++i) {
#pragma unroll
        for (int r = 0; r < 8; ++r) {
            int   co = (coT * MT + i) * 16 + r + 8 * half;
            float v  = acc[i][r];
            if (s)   v = v * s[co] + t[co];
            if (act) v = gelu_f(v);
            size_t ci = ((size_t)b * CO + co) * NPIX + p;
            if (res)   v += res[ci];
            if (outf)  outf[ci]  = v;
            if (outbc) outbc[ci] = to_bf(v);
            if (outb)  outb[((size_t)b * NPIX + p) * CBo + co] = to_bf(v);
        }
    }
}

// ---------------------------------------------------------------------------
// Depthwise 3x3 (SAME, zero pad) on bf16 ch-major input -> f32 ch-major (qkv)
// ---------------------------------------------------------------------------
__global__ void dwconv_qkv(const __bf16* __restrict__ in, const float* __restrict__ w,
                           float* __restrict__ out) {
    size_t tid = (size_t)blockIdx.x * blockDim.x + threadIdx.x;
    size_t total = (size_t)BATCH * QKVC * NPIX;
    if (tid >= total) return;
    int p = (int)(tid % NPIX);
    size_t bc = tid / NPIX;
    int c = (int)(bc % QKVC);
    int hh = p >> 7, ww = p & 127;
    const __bf16* ib = in + bc * (size_t)NPIX;
    const float* wc = w + c * 9;
    float acc = 0.f;
#pragma unroll
    for (int kh = 0; kh < 3; ++kh) {
        int nh = hh + kh - 1;
        if ((unsigned)nh >= (unsigned)HH) continue;
#pragma unroll
        for (int kw = 0; kw < 3; ++kw) {
            int nw = ww + kw - 1;
            if ((unsigned)nw >= (unsigned)WW2) continue;
            acc += wc[kh * 3 + kw] * from_bf(ib[nh * WW2 + nw]);
        }
    }
    out[tid] = acc;
}

// ---------------------------------------------------------------------------
// Depthwise 3x3 + BN + GELU: f32 ch-major [B][Cx][N] -> bf16 pixel-major [B][N][Cx]
// ---------------------------------------------------------------------------
__global__ void dw_cbn(const float* __restrict__ in, const float* __restrict__ w,
                       const float* __restrict__ s, const float* __restrict__ t,
                       __bf16* __restrict__ out, int Cx) {
    size_t tid = (size_t)blockIdx.x * blockDim.x + threadIdx.x;
    size_t total = (size_t)BATCH * Cx * NPIX;
    if (tid >= total) return;
    int p = (int)(tid % NPIX);
    size_t bc = tid / NPIX;
    int c = (int)(bc % Cx);
    size_t b = bc / Cx;
    int hh = p >> 7, ww = p & 127;
    const float* ib = in + bc * (size_t)NPIX;
    const float* wc = w + c * 9;
    float acc = 0.f;
#pragma unroll
    for (int kh = 0; kh < 3; ++kh) {
        int nh = hh + kh - 1;
        if ((unsigned)nh >= (unsigned)HH) continue;
#pragma unroll
        for (int kw = 0; kw < 3; ++kw) {
            int nw = ww + kw - 1;
            if ((unsigned)nw >= (unsigned)WW2) continue;
            acc += wc[kh * 3 + kw] * ib[nh * WW2 + nw];
        }
    }
    float v = gelu_f(acc * s[c] + t[c]);
    out[(b * (size_t)NPIX + p) * Cx + c] = to_bf(v);
}

// ---------------------------------------------------------------------------
// 1/max(||row||_2, 1e-12) over N for the q and k channels of qkv
// grid = B*384 blocks (q rows then k rows per batch), 256 threads
// ---------------------------------------------------------------------------
__global__ void row_inv_norm(const float* __restrict__ qkv, float* __restrict__ inv) {
    int r = blockIdx.x;                 // [0, B*384)
    int b = r / 384, rem = r % 384;
    int qk = rem / DIMC, c = rem % DIMC;
    const float* base = qkv + ((size_t)b * QKVC + qk * DIMC + c) * NPIX;
    float s = 0.f;
    for (int p = threadIdx.x; p < NPIX; p += blockDim.x) {
        float v = base[p];
        s += v * v;
    }
    __shared__ float red[256];
    red[threadIdx.x] = s;
    __syncthreads();
    for (int o = 128; o > 0; o >>= 1) {
        if (threadIdx.x < o) red[threadIdx.x] += red[threadIdx.x + o];
        __syncthreads();
    }
    if (threadIdx.x == 0) inv[r] = 1.0f / fmaxf(sqrtf(red[0]), 1e-12f);
}

// ---------------------------------------------------------------------------
// attn[b][h][c][d] = temp[h] * invq[c] * invk[d] * sum_n q[c][n]*k[d][n]
// One block per (b,h); LDS-tiles 32x128 slabs of q and k.
// Each thread owns 4 consecutive (c,d) pairs (same c).
// ---------------------------------------------------------------------------
__global__ void attn_score(const float* __restrict__ qkv, const float* __restrict__ inv,
                           const float* __restrict__ temp, float* __restrict__ attn) {
    int bh = blockIdx.x;
    int h = bh % HEADS, b = bh / HEADS;
    __shared__ float qs[32 * 128];
    __shared__ float ks[32 * 128];
    int tid = threadIdx.x;
    int pair0 = tid * 4;
    int c  = pair0 >> 5;
    int d0 = pair0 & 31;
    const float* qbase = qkv + ((size_t)b * QKVC + h * HD) * NPIX;
    const float* kbase = qkv + ((size_t)b * QKVC + DIMC + h * HD) * NPIX;
    float a0 = 0.f, a1 = 0.f, a2 = 0.f, a3 = 0.f;
    for (int n0 = 0; n0 < NPIX; n0 += 128) {
        __syncthreads();
#pragma unroll
        for (int i = 0; i < 16; ++i) {
            int idx = tid + i * 256;
            int cl = idx >> 7, nl = idx & 127;
            qs[idx] = qbase[(size_t)cl * NPIX + n0 + nl];
            ks[idx] = kbase[(size_t)cl * NPIX + n0 + nl];
        }
        __syncthreads();
        const float* qr = &qs[c * 128];
        for (int j = 0; j < 128; ++j) {
            float qv = qr[j];
            a0 += qv * ks[(d0 + 0) * 128 + j];
            a1 += qv * ks[(d0 + 1) * 128 + j];
            a2 += qv * ks[(d0 + 2) * 128 + j];
            a3 += qv * ks[(d0 + 3) * 128 + j];
        }
    }
    float iq = inv[b * 384 + h * HD + c];
    float tp = temp[h];
    float* arow = attn + (((size_t)bh * HD) + c) * HD;
    arow[d0 + 0] = a0 * iq * inv[b * 384 + DIMC + h * HD + d0 + 0] * tp;
    arow[d0 + 1] = a1 * iq * inv[b * 384 + DIMC + h * HD + d0 + 1] * tp;
    arow[d0 + 2] = a2 * iq * inv[b * 384 + DIMC + h * HD + d0 + 2] * tp;
    arow[d0 + 3] = a3 * iq * inv[b * 384 + DIMC + h * HD + d0 + 3] * tp;
}

// ---------------------------------------------------------------------------
// Row softmax over 32: one wave per row (wave32 shuffle reduce)
// ---------------------------------------------------------------------------
__global__ void softmax32(float* __restrict__ attn) {
    int row = blockIdx.x;
    int d = threadIdx.x;
    float x = attn[(size_t)row * 32 + d];
    float m = x;
    for (int o = 16; o > 0; o >>= 1) m = fmaxf(m, __shfl_xor(m, o, 32));
    float e = expf(x - m);
    float ssum = e;
    for (int o = 16; o > 0; o >>= 1) ssum += __shfl_xor(ssum, o, 32);
    attn[(size_t)row * 32 + d] = e / ssum;
}

// ---------------------------------------------------------------------------
// out[b][p][h*32+c] (bf16 pixel-major) = sum_d attn[b][h][c][d] * v[b][h*32+d][p]
// One block per (b,h,c).
// ---------------------------------------------------------------------------
__global__ void attn_out(const float* __restrict__ attn, const float* __restrict__ qkv,
                         __bf16* __restrict__ outb) {
    int bhc = blockIdx.x;
    int c = bhc & 31;
    int bh = bhc >> 5;
    int h = bh % HEADS, b = bh / HEADS;
    __shared__ float ar[32];
    if (threadIdx.x < 32) ar[threadIdx.x] = attn[(size_t)bhc * 32 + threadIdx.x];
    __syncthreads();
    const float* vbase = qkv + ((size_t)b * QKVC + 2 * DIMC + h * HD) * NPIX;
    for (int p = threadIdx.x; p < NPIX; p += blockDim.x) {
        float acc = 0.f;
#pragma unroll
        for (int d = 0; d < 32; ++d) acc += ar[d] * vbase[(size_t)d * NPIX + p];
        outb[((size_t)b * NPIX + p) * DIMC + h * HD + c] = to_bf(acc);
    }
}

// ---------------------------------------------------------------------------
// Spatial gate: sp[b][p] = sigmoid(bn1( sum_c spw[c]*x2f[b][c][p] + spb ))
// ---------------------------------------------------------------------------
__global__ void sp_gate(const float* __restrict__ x2f, const float* __restrict__ spw,
                        const float* __restrict__ spb, const float* __restrict__ s1,
                        const float* __restrict__ t1, float* __restrict__ sp) {
    size_t pid = (size_t)blockIdx.x * blockDim.x + threadIdx.x;
    if (pid >= (size_t)BATCH * NPIX) return;
    size_t b = pid / NPIX, p = pid % NPIX;
    const float* xb = x2f + b * DIMC * (size_t)NPIX + p;
    float acc = spb[0];
    for (int c = 0; c < DIMC; ++c) acc += spw[c] * xb[(size_t)c * NPIX];
    acc = acc * s1[0] + t1[0];
    sp[pid] = sigmoid_f(acc);
}

// ---------------------------------------------------------------------------
// Channel gate: ch[b][c] = sigmoid( mean_p( dw3x3(x1f)[b][c][p] ) + chb[c] )
// One block per (b,c).
// ---------------------------------------------------------------------------
__global__ void ch_gate(const float* __restrict__ x1f, const float* __restrict__ w,
                        const float* __restrict__ bias, float* __restrict__ ch) {
    int bc = blockIdx.x;                 // [0, B*192)
    int c = bc % DIMC;
    const float* ib = x1f + (size_t)bc * NPIX;
    const float* wc = w + c * 9;
    float acc = 0.f;
    for (int p = threadIdx.x; p < NPIX; p += blockDim.x) {
        int hh = p >> 7, ww = p & 127;
        float v = 0.f;
#pragma unroll
        for (int kh = 0; kh < 3; ++kh) {
            int nh = hh + kh - 1;
            if ((unsigned)nh >= (unsigned)HH) continue;
#pragma unroll
            for (int kw = 0; kw < 3; ++kw) {
                int nw = ww + kw - 1;
                if ((unsigned)nw >= (unsigned)WW2) continue;
                v += wc[kh * 3 + kw] * ib[nh * WW2 + nw];
            }
        }
        acc += v;
    }
    __shared__ float red[256];
    red[threadIdx.x] = acc;
    __syncthreads();
    for (int o = 128; o > 0; o >>= 1) {
        if (threadIdx.x < o) red[threadIdx.x] += red[threadIdx.x + o];
        __syncthreads();
    }
    if (threadIdx.x == 0)
        ch[bc] = sigmoid_f(red[0] * (1.0f / NPIX) + bias[c]);
}

// ---------------------------------------------------------------------------
// fuse = sp*x1 + ch*x2 + x3  ->  bf16 pixel-major [B][N][192]
// ---------------------------------------------------------------------------
__global__ void fuse_kernel(const float* __restrict__ sp, const float* __restrict__ ch,
                            const float* __restrict__ x1f, const float* __restrict__ x2f,
                            const float* __restrict__ x3f, __bf16* __restrict__ fuse) {
    size_t tid = (size_t)blockIdx.x * blockDim.x + threadIdx.x;
    size_t total = (size_t)BATCH * DIMC * NPIX;
    if (tid >= total) return;
    size_t p = tid % NPIX;
    size_t bc = tid / NPIX;
    size_t c = bc % DIMC, b = bc / DIMC;
    float v = sp[b * NPIX + p] * x1f[tid] + ch[bc] * x2f[tid] + x3f[tid];
    fuse[(b * (size_t)NPIX + p) * DIMC + c] = to_bf(v);
}

// ---------------------------------------------------------------------------
// Host-side orchestration
// ---------------------------------------------------------------------------
extern "C" void kernel_launch(void* const* d_in, const int* in_sizes, int n_in,
                              void* d_out, int out_size, void* d_ws, size_t ws_size,
                              hipStream_t stream) {
    (void)in_sizes; (void)n_in; (void)out_size; (void)ws_size;

    // ---- input pointers (setup_inputs dict order: x, then params) ----
    const float* x_in      = (const float*)d_in[0];
    const float* norm1_w   = (const float*)d_in[1];
    const float* norm1_b   = (const float*)d_in[2];
    const float* norm2_w   = (const float*)d_in[3];
    const float* norm2_b   = (const float*)d_in[4];
    const float* qkv_w     = (const float*)d_in[5];
    const float* qkv_dw_w  = (const float*)d_in[6];
    const float* temp      = (const float*)d_in[7];
    const float* aproj_w   = (const float*)d_in[8];
    const float* p1_w = (const float*)d_in[9];
    const float* p1_g = (const float*)d_in[10], *p1_bb = (const float*)d_in[11];
    const float* p1_m = (const float*)d_in[12], *p1_v  = (const float*)d_in[13];
    const float* dw1_w = (const float*)d_in[14];
    const float* dw1_g = (const float*)d_in[15], *dw1_bb = (const float*)d_in[16];
    const float* dw1_m = (const float*)d_in[17], *dw1_v  = (const float*)d_in[18];
    const float* u1_w = (const float*)d_in[19];
    const float* u1_g = (const float*)d_in[20], *u1_bb = (const float*)d_in[21];
    const float* u1_m = (const float*)d_in[22], *u1_v  = (const float*)d_in[23];
    const float* p2_w = (const float*)d_in[24];
    const float* p2_g = (const float*)d_in[25], *p2_bb = (const float*)d_in[26];
    const float* p2_m = (const float*)d_in[27], *p2_v  = (const float*)d_in[28];
    const float* dw2_w = (const float*)d_in[29];
    const float* dw2_g = (const float*)d_in[30], *dw2_bb = (const float*)d_in[31];
    const float* dw2_m = (const float*)d_in[32], *dw2_v  = (const float*)d_in[33];
    const float* u2_w = (const float*)d_in[34];
    const float* u2_g = (const float*)d_in[35], *u2_bb = (const float*)d_in[36];
    const float* u2_m = (const float*)d_in[37], *u2_v  = (const float*)d_in[38];
    const float* p3_w = (const float*)d_in[39];
    const float* p3_g = (const float*)d_in[40], *p3_bb = (const float*)d_in[41];
    const float* p3_m = (const float*)d_in[42], *p3_v  = (const float*)d_in[43];
    const float* sp_w   = (const float*)d_in[44];
    const float* sp_b   = (const float*)d_in[45];
    const float* spbn_g = (const float*)d_in[46], *spbn_b = (const float*)d_in[47];
    const float* spbn_m = (const float*)d_in[48], *spbn_v = (const float*)d_in[49];
    const float* ch_w = (const float*)d_in[50];
    const float* ch_b = (const float*)d_in[51];
    const float* fo_w = (const float*)d_in[52];
    const float* fo_g = (const float*)d_in[53], *fo_bb = (const float*)d_in[54];
    const float* fo_m = (const float*)d_in[55], *fo_v  = (const float*)d_in[56];
    float* out = (float*)d_out;

    // ---- workspace carving (aliased pools) ----
    char* ws = (char*)d_ws;
    size_t off = 0;
    auto carve = [&](size_t bytes) -> char* {
        off = (off + 255) & ~(size_t)255;
        char* p = ws + off;
        off += bytes;
        return p;
    };
    const size_t BF_PM_192 = (size_t)BATCH * NPIX * DIMC * 2;   // 50.3 MB
    // pool A: ln1-out -> attn-out -> ln2-out -> fuse (all bf16 [B][N][192])
    __bf16* poolA = (__bf16*)carve(BF_PM_192);
    // pool B: qkv_pre bf16 ch-major [B][576][N]; later h1a/h2a f32 + h1b/h2b bf16
    char* poolB = carve((size_t)BATCH * QKVC * NPIX * 2);       // 151 MB
    __bf16* qkv_pre = (__bf16*)poolB;
    float*  h1a = (float*)poolB;                                           // 50.3 MB
    float*  h2a = (float*)(poolB + (size_t)BATCH * H1C * NPIX * 4);        // 50.3 MB
    __bf16* h1b = (__bf16*)(poolB + (size_t)BATCH * H1C * NPIX * 8);       // 25.2 MB
    __bf16* h2b = (__bf16*)(poolB + (size_t)BATCH * H1C * NPIX * 8
                                  + (size_t)BATCH * NPIX * H1C * 2);       // 25.2 MB
    // pool C: qkv f32 [B][576][N]; later x1f/x2f/x3f f32 [B][192][N] each
    char* poolC = carve((size_t)BATCH * QKVC * NPIX * 4);       // 302 MB
    float* qkvf = (float*)poolC;
    float* x1f = (float*)poolC;
    float* x2f = (float*)(poolC + (size_t)BATCH * DIMC * NPIX * 4);
    float* x3f = (float*)(poolC + (size_t)BATCH * DIMC * NPIX * 8);
    // packed weights
    __bf16* Wqkv = (__bf16*)carve((QKVC/16) * (192/32) * 512 * 2);
    __bf16* Wap  = (__bf16*)carve((DIMC/16) * (192/32) * 512 * 2);
    __bf16* Wp1  = (__bf16*)carve((H1C/16)  * (64/32)  * 512 * 2);
    __bf16* Wp2  = (__bf16*)carve((H2C/16)  * (64/32)  * 512 * 2);
    __bf16* Wu1  = (__bf16*)carve((DIMC/16) * (96/32)  * 512 * 2);
    __bf16* Wu2  = (__bf16*)carve((DIMC/16) * (96/32)  * 512 * 2);
    __bf16* Wp3  = (__bf16*)carve((DIMC/16) * (96/32)  * 512 * 2);
    __bf16* Wfo  = (__bf16*)carve((DIMC/16) * (192/32) * 512 * 2);
    // folded BN consts
    float* sP1 = (float*)carve(96*4),  *tP1 = (float*)carve(96*4);
    float* sD1 = (float*)carve(96*4),  *tD1 = (float*)carve(96*4);
    float* sU1 = (float*)carve(192*4), *tU1 = (float*)carve(192*4);
    float* sP2 = (float*)carve(96*4),  *tP2 = (float*)carve(96*4);
    float* sD2 = (float*)carve(96*4),  *tD2 = (float*)carve(96*4);
    float* sU2 = (float*)carve(192*4), *tU2 = (float*)carve(192*4);
    float* sP3 = (float*)carve(192*4), *tP3 = (float*)carve(192*4);
    float* sFO = (float*)carve(192*4), *tFO = (float*)carve(192*4);
    float* sSP = (float*)carve(4),     *tSP = (float*)carve(4);
    // small activation buffers
    float* invn  = (float*)carve((size_t)BATCH * 384 * 4);
    float* attnb = (float*)carve((size_t)BATCH * HEADS * HD * HD * 4);
    float* spbuf = (float*)carve((size_t)BATCH * NPIX * 4);
    float* chbuf = (float*)carve((size_t)BATCH * DIMC * 4);

    // ---- 0. fold BN params ----
    bn_prep<<<1, 256, 0, stream>>>(p1_g,  p1_bb,  p1_m,  p1_v,  sP1, tP1, 96);
    bn_prep<<<1, 256, 0, stream>>>(dw1_g, dw1_bb, dw1_m, dw1_v, sD1, tD1, 96);
    bn_prep<<<1, 256, 0, stream>>>(u1_g,  u1_bb,  u1_m,  u1_v,  sU1, tU1, 192);
    bn_prep<<<1, 256, 0, stream>>>(p2_g,  p2_bb,  p2_m,  p2_v,  sP2, tP2, 96);
    bn_prep<<<1, 256, 0, stream>>>(dw2_g, dw2_bb, dw2_m, dw2_v, sD2, tD2, 96);
    bn_prep<<<1, 256, 0, stream>>>(u2_g,  u2_bb,  u2_m,  u2_v,  sU2, tU2, 192);
    bn_prep<<<1, 256, 0, stream>>>(p3_g,  p3_bb,  p3_m,  p3_v,  sP3, tP3, 192);
    bn_prep<<<1, 256, 0, stream>>>(fo_g,  fo_bb,  fo_m,  fo_v,  sFO, tFO, 192);
    bn_prep<<<1, 256, 0, stream>>>(spbn_g, spbn_b, spbn_m, spbn_v, sSP, tSP, 1);

    // ---- 1. pack GEMM weights into WMMA A-layout ----
    auto packs = [&](const float* W, __bf16* O, int CO, int CI, int KP) {
        int total = (CO / 16) * (KP / 32) * 512;
        pack_w<<<(total + 255) / 256, 256, 0, stream>>>(W, O, CO, CI, KP);
    };
    packs(qkv_w,   Wqkv, QKVC, 192, 192);
    packs(aproj_w, Wap,  DIMC, 192, 192);
    packs(p1_w,    Wp1,  H1C,  48,  64);
    packs(p2_w,    Wp2,  H2C,  48,  64);
    packs(u1_w,    Wu1,  DIMC, 96,  96);
    packs(u2_w,    Wu2,  DIMC, 96,  96);
    packs(p3_w,    Wp3,  DIMC, 96,  96);
    packs(fo_w,    Wfo,  DIMC, 192, 192);

    const size_t PIX_TOT = (size_t)BATCH * NPIX;
    // MT=4 when CO % 64 == 0, else MT=2 (CO % 32 == 0)
    auto gemm = [&](const __bf16* A, const __bf16* Bm, int CB, int cio, int KP,
                    int CO, const float* s, const float* t, int act,
                    const float* res, float* of, __bf16* obc, __bf16* ob, int CBo) {
        if (CO % 64 == 0) {
            dim3 grid(NPIX / 128, CO / 64, BATCH);
            gemm_wmma<4><<<grid, 256, 0, stream>>>(A, Bm, CB, cio, KP, CO,
                                                   s, t, act, res, of, obc, ob, CBo);
        } else {
            dim3 grid(NPIX / 128, CO / 32, BATCH);
            gemm_wmma<2><<<grid, 256, 0, stream>>>(A, Bm, CB, cio, KP, CO,
                                                   s, t, act, res, of, obc, ob, CBo);
        }
    };

    // ---- 2. LN1 -> bf16 pixel-major ----
    ln_kernel<<<(int)((PIX_TOT + 255) / 256), 256, 0, stream>>>(x_in, norm1_w, norm1_b, poolA);

    // ---- 3. qkv 1x1 (WMMA) -> bf16 ch-major ----
    gemm(Wqkv, poolA, DIMC, 0, 192, QKVC, nullptr, nullptr, 0,
         nullptr, nullptr, qkv_pre, nullptr, 0);

    // ---- 4. qkv depthwise 3x3 -> f32 ch-major ----
    {
        size_t total = (size_t)BATCH * QKVC * NPIX;
        dwconv_qkv<<<(int)((total + 255) / 256), 256, 0, stream>>>(qkv_pre, qkv_dw_w, qkvf);
    }

    // ---- 5. attention ----
    row_inv_norm<<<BATCH * 384, 256, 0, stream>>>(qkvf, invn);
    attn_score<<<BATCH * HEADS, 256, 0, stream>>>(qkvf, invn, temp, attnb);
    softmax32<<<BATCH * HEADS * HD, 32, 0, stream>>>(attnb);
    attn_out<<<BATCH * HEADS * HD, 256, 0, stream>>>(attnb, qkvf, poolA);

    // ---- 6. attn projection + residual (WMMA): d_out = x + proj(attn_out) ----
    gemm(Wap, poolA, DIMC, 0, 192, DIMC, nullptr, nullptr, 0,
         x_in, out, nullptr, nullptr, 0);

    // ---- 7. LN2 -> bf16 pixel-major ----
    ln_kernel<<<(int)((PIX_TOT + 255) / 256), 256, 0, stream>>>(out, norm2_w, norm2_b, poolA);

    // ---- 8. FFN branch GEMMs (WMMA, fused BN+GELU) ----
    gemm(Wp1, poolA, DIMC, 0,  64, H1C,  sP1, tP1, 1, nullptr, h1a, nullptr, nullptr, 0);
    gemm(Wp2, poolA, DIMC, 48, 64, H2C,  sP2, tP2, 1, nullptr, h2a, nullptr, nullptr, 0);
    gemm(Wp3, poolA, DIMC, 96, 96, DIMC, sP3, tP3, 0, nullptr, x3f, nullptr, nullptr, 0);

    // ---- 9. depthwise 3x3 + BN + GELU -> bf16 pixel-major ----
    {
        size_t total = (size_t)BATCH * H1C * NPIX;
        dw_cbn<<<(int)((total + 255) / 256), 256, 0, stream>>>(h1a, dw1_w, sD1, tD1, h1b, H1C);
        dw_cbn<<<(int)((total + 255) / 256), 256, 0, stream>>>(h2a, dw2_w, sD2, tD2, h2b, H2C);
    }

    // ---- 10. unify GEMMs (WMMA, fused BN+GELU) -> f32 ch-major ----
    gemm(Wu1, h1b, H1C, 0, 96, DIMC, sU1, tU1, 1, nullptr, x1f, nullptr, nullptr, 0);
    gemm(Wu2, h2b, H2C, 0, 96, DIMC, sU2, tU2, 1, nullptr, x2f, nullptr, nullptr, 0);

    // ---- 11. gates + fuse ----
    sp_gate<<<(int)((PIX_TOT + 255) / 256), 256, 0, stream>>>(x2f, sp_w, sp_b, sSP, tSP, spbuf);
    ch_gate<<<BATCH * DIMC, 256, 0, stream>>>(x1f, ch_w, ch_b, chbuf);
    {
        size_t total = (size_t)BATCH * DIMC * NPIX;
        fuse_kernel<<<(int)((total + 255) / 256), 256, 0, stream>>>(
            spbuf, chbuf, x1f, x2f, x3f, poolA);
    }

    // ---- 12. ffn_out (WMMA, fused BN) + residual into d_out ----
    gemm(Wfo, poolA, DIMC, 0, 192, DIMC, sFO, tFO, 0, out, out, nullptr, nullptr, 0);
}
